// _GraphAttnLayer_28896539968202
// MI455X (gfx1250) — compile-verified, matrix operand order checked by reading, not run on hardware
//
#include <hip/hip_runtime.h>

// ---------------------------------------------------------------------------
// Types (POD ext-vectors; HIP's uint4/float4 have ctors and can't live in unions)
// ---------------------------------------------------------------------------
typedef __attribute__((ext_vector_type(16))) __bf16 v16bf;
typedef __attribute__((ext_vector_type(8)))  float  v8f;
typedef __attribute__((ext_vector_type(4)))  unsigned int u32x4;
typedef __attribute__((ext_vector_type(4)))  float  f32x4;

union Frag {
  u32x4 q[2];   // two 16-byte chunks per lane (A/B fragment layout, 16 bf16)
  v16bf v;
};

__device__ __forceinline__ unsigned short f32_to_bf16(float f) {
  union { float f; unsigned int u; } cv; cv.f = f;
  unsigned int u = cv.u;
  if ((u & 0x7F800000u) == 0x7F800000u) return (unsigned short)(u >> 16); // Inf/NaN
  return (unsigned short)((u + 0x7FFFu + ((u >> 16) & 1u)) >> 16);        // RNE
}

__device__ __forceinline__ unsigned int pack_bf16x2(float a, float b) {
#if __has_builtin(__builtin_amdgcn_cvt_pk_bf16_f32)
  return __builtin_bit_cast(unsigned int, __builtin_amdgcn_cvt_pk_bf16_f32(a, b));
#else
  return (unsigned int)f32_to_bf16(a) | ((unsigned int)f32_to_bf16(b) << 16);
#endif
}

__device__ __forceinline__ v8f wmma_bf16f32(const Frag& a, const Frag& b, v8f c) {
  // (neg_a, A, neg_b, B, c_mod, C, reuse_a, reuse_b)
  return __builtin_amdgcn_wmma_f32_16x16x32_bf16(false, a.v, false, b.v,
                                                 (short)0, c, false, false);
}

// CDNA5 async global -> LDS copy (16B per lane), tracked on ASYNCcnt.
__device__ __forceinline__ void async_copy_b128(unsigned lds_off, const void* gaddr) {
  asm volatile("global_load_async_to_lds_b128 %0, %1, off"
               :: "v"(lds_off), "v"(gaddr) : "memory");
}

__device__ __forceinline__ void wait_async0() {
#if __has_builtin(__builtin_amdgcn_s_wait_asynccnt)
  __builtin_amdgcn_s_wait_asynccnt(0);
#else
  asm volatile("s_wait_asynccnt 0x0" ::: "memory");
#endif
}

// LDS byte offset of a __shared__ object (addrspace(3) offset = low 32 bits).
__device__ __forceinline__ unsigned lds_offset(const void* p) {
  return (unsigned)(unsigned long long)p;
}

// ---------------------------------------------------------------------------
// Problem constants
// ---------------------------------------------------------------------------
#define UU    2048
#define DD    2048
#define DFF   8192
#define NHEAD 16
#define DHEAD 128

// ---------------------------------------------------------------------------
// RMSNorm: fp32 in -> bf16 out (feeds the WMMA GEMMs)
// ---------------------------------------------------------------------------
__global__ __launch_bounds__(256)
void rmsnorm_bf16_kernel(const float* __restrict__ x, const float* __restrict__ w,
                         unsigned short* __restrict__ out, int D) {
  const int row = blockIdx.x;
  const int tid = threadIdx.x;
  const float* xr = x + (size_t)row * D;

  float ss = 0.0f;
  for (int c = tid; c < D; c += 256) { float v = xr[c]; ss += v * v; }
  for (int o = 16; o; o >>= 1) ss += __shfl_xor(ss, o, 32);

  __shared__ float red[8];
  __shared__ float s_inv;
  if ((tid & 31) == 0) red[tid >> 5] = ss;
  __syncthreads();
  if (tid == 0) {
    float tot = 0.0f;
    for (int i = 0; i < 8; ++i) tot += red[i];
    s_inv = rsqrtf(tot / (float)D + 1e-6f);
  }
  __syncthreads();
  const float inv = s_inv;
  unsigned short* orow = out + (size_t)row * D;
  for (int c = tid; c < D; c += 256)
    orow[c] = f32_to_bf16(xr[c] * inv * w[c]);
}

// ---------------------------------------------------------------------------
// Generic bf16 WMMA GEMM:  C[M,N] = A[M,K](bf16) @ Bw[N,K](f32->bf16).T
// Block tile 256x128, 8 waves as 4(M) x 2(N); wave tile 64x64 = 16 WMMA accums
// (16 WMMAs per 8 fragment loads per K-step).
// A tile staged with GLOBAL_LOAD_ASYNC_TO_LDS_B128 (ASYNCcnt path);
// B tile staged through VGPRs with fp32->bf16 packed conversion.
// ---------------------------------------------------------------------------
#define BM 256
#define BN 128
#define BK 32
#define A_STRIDE 40   // bf16 elements per LDS row (32 + 8 pad; rows stay 16B aligned)
#define B_STRIDE 40

enum { EPI_BF16 = 0, EPI_BF16_T = 1, EPI_RES_F32 = 2, EPI_GELU_BF16 = 3, EPI_RES_BIAS_F32 = 4 };

template <int EPI>
__global__ __launch_bounds__(256)
void gemm_bf16_kernel(const unsigned short* __restrict__ A,  // [M x K] bf16
                      const float* __restrict__ Bw,           // [N x K] f32 (torch Linear layout)
                      const float* __restrict__ bias,         // [N] or null
                      const float* __restrict__ resid,        // [M x N] f32 or null
                      unsigned short* __restrict__ outb,      // bf16 out or null
                      float* __restrict__ outf,               // f32 out or null
                      int M, int N, int K) {
  __shared__ __attribute__((aligned(16))) unsigned short As[BM * A_STRIDE];
  __shared__ __attribute__((aligned(16))) unsigned short Bs[BN * B_STRIDE];

  const int tid  = threadIdx.x;
  const int lane = tid & 31;
  const int wave = tid >> 5;
  const int wm   = (wave & 3) * 64;   // 4 waves along M
  const int wn   = (wave >> 2) * 64;  // 2 waves along N
  const int m0   = blockIdx.y * BM;
  const int n0   = blockIdx.x * BN;

  const int lr = lane & 15;        // fragment row (A) / col (B)
  const int hi = lane >> 4;        // half-wave select
  const int lk = hi * 8;           // K sub-base in elements

  const v8f vzero = {0.f, 0.f, 0.f, 0.f, 0.f, 0.f, 0.f, 0.f};
  v8f acc[4][4];
#pragma unroll
  for (int i = 0; i < 4; ++i)
#pragma unroll
    for (int j = 0; j < 4; ++j) acc[i][j] = vzero;

  // staging roles
  const int arow   = tid;          // 0..255 : one A row per thread (4 x 16B chunks)
  const int brow   = tid >> 1;     // 0..127 : B row
  const int bhalf  = tid & 1;      // 16 floats per thread

  const unsigned as_base = lds_offset(&As[arow * A_STRIDE]);

  for (int k0 = 0; k0 < K; k0 += BK) {
    // --- stage A tile (bf16, 256 x 32) via async DMA to LDS: 64B per thread ---
    {
      const unsigned short* src = A + (size_t)(m0 + arow) * K + k0;
#pragma unroll
      for (int c = 0; c < 4; ++c)
        async_copy_b128(as_base + c * 16, (const void*)((const char*)src + c * 16));
    }
    // --- stage B tile (f32 -> bf16, 128 x 32): cvt 16 floats per thread ---
    {
      const float* src = Bw + (size_t)(n0 + brow) * K + k0 + bhalf * 16;
      f32x4 f0 = *(const f32x4*)(src + 0);
      f32x4 f1 = *(const f32x4*)(src + 4);
      f32x4 f2 = *(const f32x4*)(src + 8);
      f32x4 f3 = *(const f32x4*)(src + 12);
      u32x4 p;
      p.x = pack_bf16x2(f0.x, f0.y);
      p.y = pack_bf16x2(f0.z, f0.w);
      p.z = pack_bf16x2(f1.x, f1.y);
      p.w = pack_bf16x2(f1.z, f1.w);
      *(u32x4*)&Bs[brow * B_STRIDE + bhalf * 16 + 0] = p;
      p.x = pack_bf16x2(f2.x, f2.y);
      p.y = pack_bf16x2(f2.z, f2.w);
      p.z = pack_bf16x2(f3.x, f3.y);
      p.w = pack_bf16x2(f3.z, f3.w);
      *(u32x4*)&Bs[brow * B_STRIDE + bhalf * 16 + 8] = p;
    }
    if (k0 + BK < K) {  // global_prefetch_b8 of next B tile (A goes via async DMA)
      __builtin_prefetch(Bw + (size_t)(n0 + brow) * K + k0 + BK, 0, 0);
    }
    wait_async0();
    __syncthreads();

    // --- fragments + 16 WMMAs per wave ---
    Frag af[4], bf[4];
#pragma unroll
    for (int i = 0; i < 4; ++i) {
      const unsigned short* p = &As[(wm + i * 16 + lr) * A_STRIDE + lk];
      af[i].q[0] = *(const u32x4*)(p);
      af[i].q[1] = *(const u32x4*)(p + 16);
    }
#pragma unroll
    for (int j = 0; j < 4; ++j) {
      const unsigned short* p = &Bs[(wn + j * 16 + lr) * B_STRIDE + lk];
      bf[j].q[0] = *(const u32x4*)(p);
      bf[j].q[1] = *(const u32x4*)(p + 16);
    }
#pragma unroll
    for (int i = 0; i < 4; ++i)
#pragma unroll
      for (int j = 0; j < 4; ++j)
        acc[i][j] = wmma_bf16f32(af[i], bf[j], acc[i][j]);
    __syncthreads();
  }

  // --- epilogue ---
#pragma unroll
  for (int i = 0; i < 4; ++i) {
#pragma unroll
    for (int j = 0; j < 4; ++j) {
#pragma unroll
      for (int r = 0; r < 8; ++r) {
        const int gm = m0 + wm + i * 16 + r + hi * 8;
        const int gn = n0 + wn + j * 16 + lr;
        float v = acc[i][j][r];
        if (EPI == EPI_BF16) {
          outb[(size_t)gm * N + gn] = f32_to_bf16(v);
        } else if (EPI == EPI_BF16_T) {
          outb[(size_t)gn * M + gm] = f32_to_bf16(v);          // store transposed (V^T)
        } else if (EPI == EPI_RES_F32) {
          outf[(size_t)gm * N + gn] = resid[(size_t)gm * N + gn] + v;
        } else if (EPI == EPI_GELU_BF16) {
          v += bias[gn];
          float g = 0.5f * v * (1.0f + erff(v * 0.70710678118654752f));
          outb[(size_t)gm * N + gn] = f32_to_bf16(g);
        } else { // EPI_RES_BIAS_F32
          outf[(size_t)gm * N + gn] = resid[(size_t)gm * N + gn] + v + bias[gn];
        }
      }
    }
  }
}

// ---------------------------------------------------------------------------
// Attention: per block = (head, 16-row strip). Scores kept in LDS (fp32),
// softmax in-place with unnormalized exps, 1/sum folded into the output.
// ---------------------------------------------------------------------------
#define SSTR (UU + 4)   // f32 per score row (pad keeps 16B row alignment: 2052*4)

__global__ __launch_bounds__(256)
void attn_kernel(const unsigned short* __restrict__ q,    // [U, D] bf16
                 const unsigned short* __restrict__ k,    // [U, D] bf16
                 const unsigned short* __restrict__ vT,   // [D, U] bf16 (V transposed)
                 const float* __restrict__ adj_bias,      // [U, U] f32
                 unsigned short* __restrict__ outb) {     // [U, D] bf16
  __shared__ __attribute__((aligned(16))) float S[16 * SSTR];  // ~128 KB
  __shared__ float inv16[16];

  const int h    = blockIdx.y;
  const int m0   = blockIdx.x * 16;
  const int tid  = threadIdx.x;
  const int lane = tid & 31;
  const int wave = tid >> 5;
  const int lr   = lane & 15;
  const int hi   = lane >> 4;
  const int lk   = hi * 8;
  const float scale = 0.08838834764831845f;  // 1/sqrt(128)

  // Q fragments for this 16-row strip: 4 frags cover d_head = 128
  Frag qf[4];
  {
    const unsigned short* qrow = q + (size_t)(m0 + lr) * DD + h * DHEAD;
#pragma unroll
    for (int i = 0; i < 4; ++i) {
      qf[i].q[0] = *(const u32x4*)(qrow + i * 32 + lk);
      qf[i].q[1] = *(const u32x4*)(qrow + i * 32 + lk + 16);
    }
  }

  // ---- scores: S[16 x 2048] = scale * Q K^T + bias ----
  for (int ct = wave; ct < UU / 16; ct += 8) {
    const int n0 = ct * 16;
    const unsigned short* krow = k + (size_t)(n0 + lr) * DD + h * DHEAD;
    v8f c = {0.f, 0.f, 0.f, 0.f, 0.f, 0.f, 0.f, 0.f};
#pragma unroll
    for (int i = 0; i < 4; ++i) {
      Frag kf;
      kf.q[0] = *(const u32x4*)(krow + i * 32 + lk);
      kf.q[1] = *(const u32x4*)(krow + i * 32 + lk + 16);
      c = wmma_bf16f32(qf[i], kf, c);
    }
    const int n = n0 + lr;
#pragma unroll
    for (int r = 0; r < 8; ++r) {
      const int m = r + hi * 8;
      S[m * SSTR + n] = c[r] * scale + adj_bias[(size_t)(m0 + m) * UU + n];
    }
  }
  __syncthreads();

  // ---- softmax (in place, unnormalized): 16 threads per row ----
  {
    const int row = tid >> 4, sub = tid & 15;
    float* srow = &S[row * SSTR];
    float mx = -3.4e38f;
    for (int c0 = sub; c0 < UU; c0 += 16) mx = fmaxf(mx, srow[c0]);
    for (int o = 8; o; o >>= 1) mx = fmaxf(mx, __shfl_xor(mx, o, 32));
    float sum = 0.0f;
    for (int c0 = sub; c0 < UU; c0 += 16) {
      float e = __expf(srow[c0] - mx);
      srow[c0] = e;
      sum += e;
    }
    for (int o = 8; o; o >>= 1) sum += __shfl_xor(sum, o, 32);
    if (sub == 0) inv16[row] = 1.0f / sum;
  }
  __syncthreads();

  // ---- O = (P V) * inv ; wave owns 16 of the 128 head dims ----
  {
    const int n0 = wave * 16;
    const unsigned short* vrow = vT + (size_t)(h * DHEAD + n0 + lr) * UU;
    const float* srow = &S[lr * SSTR];
    v8f c = {0.f, 0.f, 0.f, 0.f, 0.f, 0.f, 0.f, 0.f};
    for (int kt = 0; kt < UU / 32; ++kt) {
      const int kb = kt * 32 + lk;
      // re-pack f32 exps -> bf16 A fragment on the fly
      f32x4 a0 = *(const f32x4*)(srow + kb + 0);
      f32x4 a1 = *(const f32x4*)(srow + kb + 4);
      f32x4 a2 = *(const f32x4*)(srow + kb + 16);
      f32x4 a3 = *(const f32x4*)(srow + kb + 20);
      Frag pf, vf;
      pf.q[0].x = pack_bf16x2(a0.x, a0.y);
      pf.q[0].y = pack_bf16x2(a0.z, a0.w);
      pf.q[0].z = pack_bf16x2(a1.x, a1.y);
      pf.q[0].w = pack_bf16x2(a1.z, a1.w);
      pf.q[1].x = pack_bf16x2(a2.x, a2.y);
      pf.q[1].y = pack_bf16x2(a2.z, a2.w);
      pf.q[1].z = pack_bf16x2(a3.x, a3.y);
      pf.q[1].w = pack_bf16x2(a3.z, a3.w);
      vf.q[0] = *(const u32x4*)(vrow + kt * 32 + lk);
      vf.q[1] = *(const u32x4*)(vrow + kt * 32 + lk + 16);
      c = wmma_bf16f32(pf, vf, c);
    }
#pragma unroll
    for (int r = 0; r < 8; ++r) {
      const int m = r + hi * 8;
      outb[(size_t)(m0 + m) * DD + h * DHEAD + n0 + lr] = f32_to_bf16(c[r] * inv16[m]);
    }
  }
}

// ---------------------------------------------------------------------------
// Host orchestration
// ---------------------------------------------------------------------------
extern "C" void kernel_launch(void* const* d_in, const int* in_sizes, int n_in,
                              void* d_out, int out_size, void* d_ws, size_t ws_size,
                              hipStream_t stream) {
  (void)in_sizes; (void)n_in; (void)out_size; (void)ws_size;

  const float* x      = (const float*)d_in[0];
  const float* adjb   = (const float*)d_in[1];
  const float* n1w    = (const float*)d_in[2];
  const float* n2w    = (const float*)d_in[3];
  const float* wq     = (const float*)d_in[4];
  const float* wk     = (const float*)d_in[5];
  const float* wv     = (const float*)d_in[6];
  const float* wo     = (const float*)d_in[7];
  const float* w_up   = (const float*)d_in[8];
  const float* b_up   = (const float*)d_in[9];
  const float* w_down = (const float*)d_in[10];
  const float* b_down = (const float*)d_in[11];
  float* out = (float*)d_out;

  char* ws = (char*)d_ws;
  const size_t MB = 1ull << 20;
  unsigned short* hb   = (unsigned short*)(ws + 0 * MB);    // [U,D] bf16   8MB
  unsigned short* qb   = (unsigned short*)(ws + 8 * MB);    // [U,D] bf16   8MB
  unsigned short* kb   = (unsigned short*)(ws + 16 * MB);   // [U,D] bf16   8MB
  unsigned short* vTb  = (unsigned short*)(ws + 24 * MB);   // [D,U] bf16   8MB
  unsigned short* aob  = (unsigned short*)(ws + 32 * MB);   // [U,D] bf16   8MB
  unsigned short* h2b  = (unsigned short*)(ws + 40 * MB);   // [U,D] bf16   8MB
  unsigned short* ffb  = (unsigned short*)(ws + 48 * MB);   // [U,DFF] bf16 32MB
  float*          xmid = (float*)        (ws + 80 * MB);    // [U,D] f32    16MB

  const dim3 blk(256, 1, 1);
  const dim3 gD(DD / BN, UU / BM, 1);      // (16, 8)
  const dim3 gFF(DFF / BN, UU / BM, 1);    // (64, 8)
  const dim3 gAttn(UU / 16, NHEAD, 1);     // (128,16)

  // h = rmsnorm(x) -> bf16
  rmsnorm_bf16_kernel<<<UU, blk, 0, stream>>>(x, n1w, hb, DD);
  // q, k, v projections (v stored transposed)
  gemm_bf16_kernel<EPI_BF16>  <<<gD, blk, 0, stream>>>(hb, wq, nullptr, nullptr, qb,  nullptr, UU, DD, DD);
  gemm_bf16_kernel<EPI_BF16>  <<<gD, blk, 0, stream>>>(hb, wk, nullptr, nullptr, kb,  nullptr, UU, DD, DD);
  gemm_bf16_kernel<EPI_BF16_T><<<gD, blk, 0, stream>>>(hb, wv, nullptr, nullptr, vTb, nullptr, UU, DD, DD);
  // attention
  attn_kernel<<<gAttn, blk, 0, stream>>>(qb, kb, vTb, adjb, aob);
  // x = x + attn_out @ wo.T
  gemm_bf16_kernel<EPI_RES_F32><<<gD, blk, 0, stream>>>(aob, wo, nullptr, x, nullptr, xmid, UU, DD, DD);
  // h2 = rmsnorm(x) -> bf16
  rmsnorm_bf16_kernel<<<UU, blk, 0, stream>>>(xmid, n2w, h2b, DD);
  // ff = gelu(h2 @ w_up.T + b_up) -> bf16
  gemm_bf16_kernel<EPI_GELU_BF16><<<gFF, blk, 0, stream>>>(h2b, w_up, b_up, nullptr, ffb, nullptr, UU, DFF, DD);
  // out = x + ff @ w_down.T + b_down
  gemm_bf16_kernel<EPI_RES_BIAS_F32><<<gD, blk, 0, stream>>>(ffb, w_down, b_down, xmid, nullptr, out, UU, DD, DFF);
}
